// GMMEqualVarianceDenoiser_84859963834485
// MI455X (gfx1250) — matrix operand order repaired and checked by reading, not compiled, not run on hardware
//
#include <hip/hip_runtime.h>
#include <hip/hip_bf16.h>
#include <math.h>

// ---------------------------------------------------------------------------
// GMM equal-variance denoiser, fused flash-softmax formulation for gfx1250.
//   Q = x [16384 x 64], K = V = scale*means [8192 x 64]
//   logits = (Q.K^T - 0.5*||K||^2) / var      (||x||^2 cancels in softmax)
//   out    = (dw*x + (1-dw)*softmax(logits) @ K) / scale
//
// Pass 1 (prep): scale means, convert to bf16 (row-major + transposed),
//                compute column bias -0.5*||c||^2/var. (~2MB, stays in L2)
// Pass 2 (main): flash kernel; double-buffered ASYNC global->LDS tile copies
//                overlap the bf16 WMMA + online-softmax compute.
// ---------------------------------------------------------------------------

typedef __attribute__((ext_vector_type(16))) __bf16 v16bf;
typedef __attribute__((ext_vector_type(8)))  __bf16 v8bf;
typedef __attribute__((ext_vector_type(8)))  float  v8f;

#define WMMA_BF16(a, b, c) \
  __builtin_amdgcn_wmma_f32_16x16x32_bf16(false, (a), false, (b), (short)0, (c), false, false)

constexpr int BATCH   = 16384;
constexpr int DIM     = 64;     // feature dim (C*H*W)
constexpr int NCLUS   = 8192;   // clusters
constexpr int NTILE   = 64;     // clusters per K/V tile
constexpr int NT      = NCLUS / NTILE;  // 128 tiles
constexpr int WAVES   = 4;      // wave32s per block
constexpr int BLOCK_M = WAVES * 16;

// workspace layout (bytes)
constexpr size_t WS_MBF_OFF  = 0;                                   // bf16 [N][D]
constexpr size_t WS_MT_OFF   = (size_t)NCLUS * DIM * 2;             // bf16 [D][N]
constexpr size_t WS_BIAS_OFF = WS_MT_OFF * 2;                       // f32  [N]
constexpr size_t WS_NEED     = WS_BIAS_OFF + (size_t)NCLUS * 4;

// ===========================================================================
// Pass 1: precompute scaled bf16 means (row-major + transposed) and bias.
// ===========================================================================
__global__ __launch_bounds__(256)
void gmm_prep_kernel(const float* __restrict__ means,
                     const float* __restrict__ t,
                     const float* __restrict__ sdv,
                     unsigned short* __restrict__ mBFu,
                     unsigned short* __restrict__ mTu,
                     float* __restrict__ bias) {
  const int c = blockIdx.x * blockDim.x + threadIdx.x;
  if (c >= NCLUS) return;
  __bf16* mBF = (__bf16*)mBFu;
  __bf16* mT  = (__bf16*)mTu;

  const float scale   = __expf(-t[0]);
  const float sd      = sdv[0];
  const float s2sd2   = scale * scale * sd * sd;
  const float var     = s2sd2 + (1.0f - scale * scale);
  const float inv_var = 1.0f / var;

  const float* src = means + (size_t)c * DIM;
  float ssum = 0.0f;
#pragma unroll
  for (int d0 = 0; d0 < DIM; d0 += 4) {
    const float4 v = *(const float4*)(src + d0);
    const __bf16 b0 = (__bf16)(v.x * scale);
    const __bf16 b1 = (__bf16)(v.y * scale);
    const __bf16 b2 = (__bf16)(v.z * scale);
    const __bf16 b3 = (__bf16)(v.w * scale);
    // bias from the rounded values (consistent with the bf16 cross terms)
    const float f0 = (float)b0, f1 = (float)b1, f2 = (float)b2, f3 = (float)b3;
    ssum += f0 * f0 + f1 * f1 + f2 * f2 + f3 * f3;
    mBF[(size_t)c * DIM + d0 + 0] = b0;
    mBF[(size_t)c * DIM + d0 + 1] = b1;
    mBF[(size_t)c * DIM + d0 + 2] = b2;
    mBF[(size_t)c * DIM + d0 + 3] = b3;
    mT[(size_t)(d0 + 0) * NCLUS + c] = b0;
    mT[(size_t)(d0 + 1) * NCLUS + c] = b1;
    mT[(size_t)(d0 + 2) * NCLUS + c] = b2;
    mT[(size_t)(d0 + 3) * NCLUS + c] = b3;
  }
  bias[c] = -0.5f * ssum * inv_var;
}

// ===========================================================================
// Pass 2: fused flash kernel with async double-buffered K/V tiles.
// ===========================================================================
__global__ __launch_bounds__(WAVES * 32)
void gmm_denoise_async(const float* __restrict__ x,
                       const float* __restrict__ t,
                       const float* __restrict__ sdv,
                       const unsigned short* __restrict__ mBFu,
                       const unsigned short* __restrict__ mTu,
                       const float* __restrict__ biasArr,
                       float* __restrict__ out) {
  __shared__ alignas(32) __bf16 sTile[2][NTILE * DIM];   // scaled bf16, [n][d]
  __shared__ alignas(32) __bf16 sTileT[2][DIM * NTILE];  // transposed, [d][n]
  __shared__ alignas(32) __bf16 sP[WAVES][16 * NTILE];   // per-wave P staging

  const __bf16* mBF = (const __bf16*)mBFu;
  const __bf16* mT  = (const __bf16*)mTu;

  const int tid     = threadIdx.x;
  const int lane    = tid & 31;
  const int laneLo  = lane & 15;
  const int halfSel = (lane >> 4) & 1;
  const int wave    = tid >> 5;

  const float tval      = t[0];
  const float sd        = sdv[0];
  const float scale     = __expf(-tval);
  const float s2sd2     = scale * scale * sd * sd;
  const float var       = s2sd2 + (1.0f - scale * scale);
  const float inv_var   = 1.0f / var;
  const float dw        = s2sd2 * inv_var;
  const float inv_scale = 1.0f / scale;

  const int rowBase = blockIdx.x * BLOCK_M + wave * 16;
  const int myRow   = rowBase + laneLo;
  const float* xrow = x + (size_t)myRow * DIM;

  // ---- load this wave's 16x64 Q tile once, in WMMA A layout (bf16) --------
  v16bf Xa[2];
#pragma unroll
  for (int s = 0; s < 2; ++s) {
    const int kb = 32 * s;
    const int cA = kb + 16 * halfSel;
    const int cB = kb + 16 * halfSel + 8;
    float4 a0 = *(const float4*)(xrow + cA);
    float4 a1 = *(const float4*)(xrow + cA + 4);
    float4 b0 = *(const float4*)(xrow + cB);
    float4 b1 = *(const float4*)(xrow + cB + 4);
    v16bf v;
    v[0]  = (__bf16)a0.x; v[1]  = (__bf16)a0.y; v[2]  = (__bf16)a0.z; v[3]  = (__bf16)a0.w;
    v[4]  = (__bf16)a1.x; v[5]  = (__bf16)a1.y; v[6]  = (__bf16)a1.z; v[7]  = (__bf16)a1.w;
    v[8]  = (__bf16)b0.x; v[9]  = (__bf16)b0.y; v[10] = (__bf16)b0.z; v[11] = (__bf16)b0.w;
    v[12] = (__bf16)b1.x; v[13] = (__bf16)b1.y; v[14] = (__bf16)b1.z; v[15] = (__bf16)b1.w;
    Xa[s] = v;
  }

  // Issue one tile's async copies: 8 x b128 per wave (4KB), 16KB per block.
  // Generic pointers to __shared__ carry the LDS offset in their low 32 bits.
  auto issueTile = [&](int buf, int tileIdx) {
    const int nBase = tileIdx * NTILE;
    const unsigned long long sA =
        (unsigned long long)(const void*)(mBF + (size_t)nBase * DIM);
    const unsigned long long sB =
        (unsigned long long)(const void*)(mT + nBase);
#pragma unroll
    for (int q = 0; q < 4; ++q) {
      const unsigned lin = (unsigned)(tid + q * 128) * 8u;  // half index in 64x64
      const unsigned ldsA = (unsigned)(uintptr_t)&sTile[buf][lin];
      asm volatile("global_load_async_to_lds_b128 %0, %1, %2"
                   :: "v"(ldsA), "v"(lin * 2u), "s"(sA) : "memory");
      const unsigned d   = lin >> 6;
      const unsigned col = lin & 63u;
      const unsigned ldsB = (unsigned)(uintptr_t)&sTileT[buf][lin];
      asm volatile("global_load_async_to_lds_b128 %0, %1, %2"
                   :: "v"(ldsB), "v"((d * (unsigned)NCLUS + col) * 2u), "s"(sB)
                   : "memory");
    }
  };

  // flash-softmax running state (C layout: VGPR r -> row r + 8*halfSel)
  v8f acc[4];
  const v8f vzero = {0, 0, 0, 0, 0, 0, 0, 0};
#pragma unroll
  for (int d = 0; d < 4; ++d) acc[d] = vzero;
  float mrow[8], lrow[8];
#pragma unroll
  for (int r = 0; r < 8; ++r) { mrow[r] = -INFINITY; lrow[r] = 0.0f; }

  issueTile(0, 0);  // prologue: tile 0 in flight

  for (int tile = 0; tile < NT; ++tile) {
    const int buf   = tile & 1;
    const int nBase = tile * NTILE;

    if (tile + 1 < NT) {
      issueTile(buf ^ 1, tile + 1);                       // prefetch next tile
      asm volatile("s_wait_asynccnt 0x8" ::: "memory");   // current tile landed
    } else {
      asm volatile("s_wait_asynccnt 0x0" ::: "memory");
    }
    __syncthreads();  // everyone's copies visible

    // ---- GEMM1: S = Q . K^T  (16 x 64 per wave), then logits ---------------
    float W[4][8];
#pragma unroll
    for (int j = 0; j < 4; ++j) {
      const int clus = j * 16 + laneLo;  // B 32x16: lane = column (cluster)
      v16bf kb0 = *(const v16bf*)&sTile[buf][clus * DIM + 16 * halfSel];
      v16bf kb1 = *(const v16bf*)&sTile[buf][clus * DIM + 32 + 16 * halfSel];
      v8f sacc = vzero;
      sacc = WMMA_BF16(Xa[0], kb0, sacc);
      sacc = WMMA_BF16(Xa[1], kb1, sacc);
      const float bj = biasArr[nBase + j * 16 + laneLo];  // L2-resident
#pragma unroll
      for (int r = 0; r < 8; ++r) W[j][r] = sacc[r] * inv_var + bj;
    }

    // ---- online softmax (row reductions live in 16-lane halves) ------------
    float nm[8];
#pragma unroll
    for (int r = 0; r < 8; ++r)
      nm[r] = fmaxf(fmaxf(W[0][r], W[1][r]), fmaxf(W[2][r], W[3][r]));
#pragma unroll
    for (int r = 0; r < 8; ++r) {
      nm[r] = fmaxf(nm[r], __shfl_xor(nm[r], 1, 16));
      nm[r] = fmaxf(nm[r], __shfl_xor(nm[r], 2, 16));
      nm[r] = fmaxf(nm[r], __shfl_xor(nm[r], 4, 16));
      nm[r] = fmaxf(nm[r], __shfl_xor(nm[r], 8, 16));
      nm[r] = fmaxf(nm[r], mrow[r]);
    }
    float rs[8];
#pragma unroll
    for (int r = 0; r < 8; ++r) {
      const float alpha = __expf(mrow[r] - nm[r]);
      mrow[r] = nm[r];
      float s0 = 0.0f;
#pragma unroll
      for (int j = 0; j < 4; ++j) {
        const float p = __expf(W[j][r] - nm[r]);
        W[j][r] = p;
        s0 += p;
      }
      rs[r] = s0;
#pragma unroll
      for (int d = 0; d < 4; ++d) acc[d][r] *= alpha;
      lrow[r] *= alpha;
    }
#pragma unroll
    for (int r = 0; r < 8; ++r) {
      rs[r] += __shfl_xor(rs[r], 1, 16);
      rs[r] += __shfl_xor(rs[r], 2, 16);
      rs[r] += __shfl_xor(rs[r], 4, 16);
      rs[r] += __shfl_xor(rs[r], 8, 16);
      lrow[r] += rs[r];
    }

    // ---- stage P (C layout -> LDS row-major) and reload in A layout --------
    __bf16* pbuf = &sP[wave][0];
#pragma unroll
    for (int j = 0; j < 4; ++j)
#pragma unroll
      for (int r = 0; r < 8; ++r)
        pbuf[(r + 8 * halfSel) * NTILE + j * 16 + laneLo] = (__bf16)W[j][r];
    asm volatile("s_wait_dscnt 0" ::: "memory");  // wave-internal LDS RAW fence

    v16bf Pa[2];
#pragma unroll
    for (int s = 0; s < 2; ++s) {
      const int kb = 32 * s;
      v8bf lo = *(const v8bf*)&pbuf[laneLo * NTILE + kb + 16 * halfSel];
      v8bf hi = *(const v8bf*)&pbuf[laneLo * NTILE + kb + 16 * halfSel + 8];
      Pa[s] = __builtin_shufflevector(lo, hi, 0, 1, 2, 3, 4, 5, 6, 7,
                                      8, 9, 10, 11, 12, 13, 14, 15);
    }

    // ---- GEMM2: O += P . V  (K dim = 64 clusters of this tile) -------------
#pragma unroll
    for (int d = 0; d < 4; ++d) {
      const int dimc = d * 16 + laneLo;  // B 32x16: lane = column (feature dim)
      v16bf vb0 = *(const v16bf*)&sTileT[buf][dimc * NTILE + 16 * halfSel];
      v16bf vb1 = *(const v16bf*)&sTileT[buf][dimc * NTILE + 32 + 16 * halfSel];
      acc[d] = WMMA_BF16(Pa[0], vb0, acc[d]);
      acc[d] = WMMA_BF16(Pa[1], vb1, acc[d]);
    }
    __syncthreads();  // tile fully consumed before next overwrite
  }

  // ---- epilogue: normalize, blend with x, undo OU scale --------------------
#pragma unroll
  for (int d = 0; d < 4; ++d) {
    const int col = d * 16 + laneLo;
#pragma unroll
    for (int r = 0; r < 8; ++r) {
      const int row = rowBase + r + 8 * halfSel;
      const float ct = acc[d][r] / lrow[r];
      const float xv = x[(size_t)row * DIM + col];
      out[(size_t)row * DIM + col] = (dw * xv + (1.0f - dw) * ct) * inv_scale;
    }
  }
}

// ===========================================================================
// Fallback (round-1 kernel): used only if the workspace is too small.
// ===========================================================================
__global__ __launch_bounds__(WAVES * 32)
void gmm_denoise_fallback(const float* __restrict__ x,
                          const float* __restrict__ t,
                          const float* __restrict__ means,
                          const float* __restrict__ sdv,
                          float* __restrict__ out) {
  __shared__ alignas(32) __bf16 sTile[NTILE * DIM];
  __shared__ alignas(32) __bf16 sTileT[DIM * NTILE];
  __shared__ alignas(32) __bf16 sP[WAVES][16 * NTILE];
  __shared__ float sBias[NTILE];

  const int tid     = threadIdx.x;
  const int lane    = tid & 31;
  const int laneLo  = lane & 15;
  const int halfSel = (lane >> 4) & 1;
  const int wave    = tid >> 5;

  const float tval      = t[0];
  const float sd        = sdv[0];
  const float scale     = __expf(-tval);
  const float s2sd2     = scale * scale * sd * sd;
  const float var       = s2sd2 + (1.0f - scale * scale);
  const float inv_var   = 1.0f / var;
  const float dw        = s2sd2 * inv_var;
  const float inv_scale = 1.0f / scale;

  const int rowBase = blockIdx.x * BLOCK_M + wave * 16;
  const float* xrow = x + (size_t)(rowBase + laneLo) * DIM;

  v16bf Xa[2];
#pragma unroll
  for (int s = 0; s < 2; ++s) {
    const int kb = 32 * s;
    const int cA = kb + 16 * halfSel;
    const int cB = kb + 16 * halfSel + 8;
    float4 a0 = *(const float4*)(xrow + cA);
    float4 a1 = *(const float4*)(xrow + cA + 4);
    float4 b0 = *(const float4*)(xrow + cB);
    float4 b1 = *(const float4*)(xrow + cB + 4);
    v16bf v;
    v[0]  = (__bf16)a0.x; v[1]  = (__bf16)a0.y; v[2]  = (__bf16)a0.z; v[3]  = (__bf16)a0.w;
    v[4]  = (__bf16)a1.x; v[5]  = (__bf16)a1.y; v[6]  = (__bf16)a1.z; v[7]  = (__bf16)a1.w;
    v[8]  = (__bf16)b0.x; v[9]  = (__bf16)b0.y; v[10] = (__bf16)b0.z; v[11] = (__bf16)b0.w;
    v[12] = (__bf16)b1.x; v[13] = (__bf16)b1.y; v[14] = (__bf16)b1.z; v[15] = (__bf16)b1.w;
    Xa[s] = v;
  }

  v8f acc[4];
  const v8f vzero = {0, 0, 0, 0, 0, 0, 0, 0};
#pragma unroll
  for (int d = 0; d < 4; ++d) acc[d] = vzero;
  float mrow[8], lrow[8];
#pragma unroll
  for (int r = 0; r < 8; ++r) { mrow[r] = -INFINITY; lrow[r] = 0.0f; }

  for (int tile = 0; tile < NT; ++tile) {
    const int nBase = tile * NTILE;
    __syncthreads();
#pragma unroll
    for (int i = 0; i < (NTILE * DIM / 4) / (WAVES * 32); ++i) {
      const int f4 = tid + i * (WAVES * 32);
      const int c  = f4 >> 4;
      const int d0 = (f4 & 15) << 2;
      float4 mv = *(const float4*)(means + (size_t)(nBase + c) * DIM + d0);
      __bf16 b0 = (__bf16)(mv.x * scale);
      __bf16 b1 = (__bf16)(mv.y * scale);
      __bf16 b2 = (__bf16)(mv.z * scale);
      __bf16 b3 = (__bf16)(mv.w * scale);
      __bf16* pt = &sTile[c * DIM + d0];
      pt[0] = b0; pt[1] = b1; pt[2] = b2; pt[3] = b3;
      sTileT[(d0 + 0) * NTILE + c] = b0;
      sTileT[(d0 + 1) * NTILE + c] = b1;
      sTileT[(d0 + 2) * NTILE + c] = b2;
      sTileT[(d0 + 3) * NTILE + c] = b3;
    }
    __syncthreads();
    if (tid < NTILE) {
      float ssum = 0.0f;
      const __bf16* rowp = &sTile[tid * DIM];
#pragma unroll
      for (int d = 0; d < DIM; ++d) { const float f = (float)rowp[d]; ssum += f * f; }
      sBias[tid] = -0.5f * ssum * inv_var;
    }
    __syncthreads();

    float W[4][8];
#pragma unroll
    for (int j = 0; j < 4; ++j) {
      const int clus = j * 16 + laneLo;
      v16bf kb0 = *(const v16bf*)&sTile[clus * DIM + 16 * halfSel];
      v16bf kb1 = *(const v16bf*)&sTile[clus * DIM + 32 + 16 * halfSel];
      v8f sacc = vzero;
      sacc = WMMA_BF16(Xa[0], kb0, sacc);
      sacc = WMMA_BF16(Xa[1], kb1, sacc);
      const float bj = sBias[j * 16 + laneLo];
#pragma unroll
      for (int r = 0; r < 8; ++r) W[j][r] = sacc[r] * inv_var + bj;
    }

    float nm[8];
#pragma unroll
    for (int r = 0; r < 8; ++r)
      nm[r] = fmaxf(fmaxf(W[0][r], W[1][r]), fmaxf(W[2][r], W[3][r]));
#pragma unroll
    for (int r = 0; r < 8; ++r) {
      nm[r] = fmaxf(nm[r], __shfl_xor(nm[r], 1, 16));
      nm[r] = fmaxf(nm[r], __shfl_xor(nm[r], 2, 16));
      nm[r] = fmaxf(nm[r], __shfl_xor(nm[r], 4, 16));
      nm[r] = fmaxf(nm[r], __shfl_xor(nm[r], 8, 16));
      nm[r] = fmaxf(nm[r], mrow[r]);
    }
    float rs[8];
#pragma unroll
    for (int r = 0; r < 8; ++r) {
      const float alpha = __expf(mrow[r] - nm[r]);
      mrow[r] = nm[r];
      float s0 = 0.0f;
#pragma unroll
      for (int j = 0; j < 4; ++j) {
        const float p = __expf(W[j][r] - nm[r]);
        W[j][r] = p;
        s0 += p;
      }
      rs[r] = s0;
#pragma unroll
      for (int d = 0; d < 4; ++d) acc[d][r] *= alpha;
      lrow[r] *= alpha;
    }
#pragma unroll
    for (int r = 0; r < 8; ++r) {
      rs[r] += __shfl_xor(rs[r], 1, 16);
      rs[r] += __shfl_xor(rs[r], 2, 16);
      rs[r] += __shfl_xor(rs[r], 4, 16);
      rs[r] += __shfl_xor(rs[r], 8, 16);
      lrow[r] += rs[r];
    }

    __bf16* pbuf = &sP[wave][0];
#pragma unroll
    for (int j = 0; j < 4; ++j)
#pragma unroll
      for (int r = 0; r < 8; ++r)
        pbuf[(r + 8 * halfSel) * NTILE + j * 16 + laneLo] = (__bf16)W[j][r];
    asm volatile("s_wait_dscnt 0" ::: "memory");

    v16bf Pa[2];
#pragma unroll
    for (int s = 0; s < 2; ++s) {
      const int kb = 32 * s;
      v8bf lo = *(const v8bf*)&pbuf[laneLo * NTILE + kb + 16 * halfSel];
      v8bf hi = *(const v8bf*)&pbuf[laneLo * NTILE + kb + 16 * halfSel + 8];
      Pa[s] = __builtin_shufflevector(lo, hi, 0, 1, 2, 3, 4, 5, 6, 7,
                                      8, 9, 10, 11, 12, 13, 14, 15);
    }

#pragma unroll
    for (int d = 0; d < 4; ++d) {
      const int dimc = d * 16 + laneLo;
      v16bf vb0 = *(const v16bf*)&sTileT[dimc * NTILE + 16 * halfSel];
      v16bf vb1 = *(const v16bf*)&sTileT[dimc * NTILE + 32 + 16 * halfSel];
      acc[d] = WMMA_BF16(Pa[0], vb0, acc[d]);
      acc[d] = WMMA_BF16(Pa[1], vb1, acc[d]);
    }
  }

#pragma unroll
  for (int d = 0; d < 4; ++d) {
    const int col = d * 16 + laneLo;
#pragma unroll
    for (int r = 0; r < 8; ++r) {
      const int row = rowBase + r + 8 * halfSel;
      const float ct = acc[d][r] / lrow[r];
      const float xv = x[(size_t)row * DIM + col];
      out[(size_t)row * DIM + col] = (dw * xv + (1.0f - dw) * ct) * inv_scale;
    }
  }
}

// ===========================================================================
extern "C" void kernel_launch(void* const* d_in, const int* in_sizes, int n_in,
                              void* d_out, int out_size, void* d_ws, size_t ws_size,
                              hipStream_t stream) {
  (void)in_sizes; (void)n_in; (void)out_size;
  const float* x     = (const float*)d_in[0];
  const float* t     = (const float*)d_in[1];
  const float* means = (const float*)d_in[2];
  const float* sd    = (const float*)d_in[3];
  float* out         = (float*)d_out;

  dim3 grid(BATCH / BLOCK_M);   // 256 blocks
  dim3 block(WAVES * 32);       // 128 threads = 4 wave32

  if (ws_size >= WS_NEED && d_ws != nullptr) {
    unsigned short* mBF = (unsigned short*)((char*)d_ws + WS_MBF_OFF);
    unsigned short* mT  = (unsigned short*)((char*)d_ws + WS_MT_OFF);
    float*          bb  = (float*)((char*)d_ws + WS_BIAS_OFF);
    hipLaunchKernelGGL(gmm_prep_kernel, dim3(NCLUS / 256), dim3(256), 0, stream,
                       means, t, sd, mBF, mT, bb);
    hipLaunchKernelGGL(gmm_denoise_async, grid, block, 0, stream,
                       x, t, sd, mBF, mT, bb, out);
  } else {
    hipLaunchKernelGGL(gmm_denoise_fallback, grid, block, 0, stream,
                       x, t, means, sd, out);
  }
}
